// GroupedQueryAttention_70153995812975
// MI455X (gfx1250) — compile-verified, matrix operand order checked by reading, not run on hardware
//
#include <hip/hip_runtime.h>
#include <hip/hip_bf16.h>

// ---------------------------------------------------------------------------
// GQA forward for MI455X (gfx1250, wave32, WMMA + Tensor Data Mover).
// All matmuls use v_wmma_f32_16x16x32_bf16 with bf16x3 (hi/lo split) operands
// for near-fp32 accuracy at bf16 matrix-pipe speed. Flash-attention K/V tiles
// are staged by TDM (tensor_load_to_lds) with double-buffered LDS.
// ---------------------------------------------------------------------------

typedef __bf16 bf16_t;
typedef __attribute__((ext_vector_type(16))) __bf16 v16bf;
typedef __attribute__((ext_vector_type(8)))  __bf16 v8bf;
typedef __attribute__((ext_vector_type(8)))  float  v8f;
typedef unsigned v4u_t __attribute__((ext_vector_type(4)));
typedef int      v4i_t __attribute__((ext_vector_type(4)));
typedef int      v8i_t __attribute__((ext_vector_type(8)));

#define HIDDEN 2048
#define NH     16
#define NKV    4
#define HD     128
#define SEQ    2048
#define BATCH  2
#define ROWS   (BATCH * SEQ)            // 4096
#define NQKV   (NH * HD + 2 * NKV * HD) // 3072

// Fragment load: lane holds 16 bf16 = chunks {k..k+7, k+16..k+23} (A/B layout,
// lane>=16 adds +8 to k which the caller folds in via kh8).
__device__ __forceinline__ v16bf frag_ld(const bf16_t* p) {
  v8bf a = *(const v8bf*)p;
  v8bf b = *(const v8bf*)(p + 16);
  v16bf r;
#pragma unroll
  for (int i = 0; i < 8; ++i) { r[i] = a[i]; r[i + 8] = b[i]; }
  return r;
}

// bf16x3 compensated MMA: (Ah+Al)(Bh+Bl) ~= AhBh + AlBh + AhBl
__device__ __forceinline__ void mma3(v8f& c, v16bf ah, v16bf al, v16bf bh, v16bf bl) {
  c = __builtin_amdgcn_wmma_f32_16x16x32_bf16(false, ah, false, bh, (short)0, c, false, false);
  c = __builtin_amdgcn_wmma_f32_16x16x32_bf16(false, al, false, bh, (short)0, c, false, false);
  c = __builtin_amdgcn_wmma_f32_16x16x32_bf16(false, ah, false, bl, (short)0, c, false, false);
}

__device__ __forceinline__ void split_store(bf16_t* H, bf16_t* L, size_t i, float v) {
  bf16_t h = (bf16_t)v;
  H[i] = h;
  L[i] = (bf16_t)(v - (float)h);
}

// LDS byte offset of a __shared__ object: flat address = {aperture, offset}
// (ISA 10.2: the low 32 bits are the LDS address).
__device__ __forceinline__ unsigned lds_off(const void* p) {
  return (unsigned)(unsigned long long)p;
}

// ---------------- Tensor Data Mover: 2-D tile load with LDS padding ---------
// Builds D# group0/group1 per CDNA5 ISA 8.3/8.4 and issues TENSOR_LOAD_TO_LDS.
// g1w0 packs {data_size=2B, pad_enable, pad_interval, pad_amount}.
__device__ __forceinline__ void tdm_load2d(unsigned lds_addr, unsigned long long gaddr,
                                           unsigned g1w0, unsigned tdim0, unsigned tdim1,
                                           unsigned tile0, unsigned tile1,
                                           unsigned stride0) {
  v4u_t g0;
  g0[0] = 1u;                                   // count=1 (valid), user mode
  g0[1] = lds_addr;                             // lds_addr [63:32]
  g0[2] = (unsigned)gaddr;                      // global_addr lo
  g0[3] = (unsigned)((gaddr >> 32) & 0x01ffffffull) | (2u << 30); // addr hi | type=2
  v8i_t g1;
  g1[0] = (int)g1w0;                                            // mask/dsize/pad
  g1[1] = (int)((tdim0 & 0xffffu) << 16);                       // tensor_dim0 lo16
  g1[2] = (int)((tdim0 >> 16) | ((tdim1 & 0xffffu) << 16));     // dim0 hi / dim1 lo
  g1[3] = (int)((tdim1 >> 16) | (tile0 << 16));                 // dim1 hi / tile0
  g1[4] = (int)tile1;                                           // tile1, tile2=0
  g1[5] = (int)stride0;                                         // dim0_stride lo
  g1[6] = 0;                                                    // stride hi / d1s lo
  g1[7] = 0;
  v4i_t z4;
  z4[0] = 0; z4[1] = 0; z4[2] = 0; z4[3] = 0;
#if __has_builtin(__builtin_amdgcn_tensor_load_to_lds)
#if __clang_major__ >= 23
  v8i_t z8;
#pragma unroll
  for (int i = 0; i < 8; ++i) z8[i] = 0;
  __builtin_amdgcn_tensor_load_to_lds(g0, g1, z4, z4, z8, 0);
#else
  __builtin_amdgcn_tensor_load_to_lds(g0, g1, z4, z4, 0);
#endif
#else
  asm volatile("tensor_load_to_lds %0, %1, %2, %3"
               :: "s"(g0), "s"(g1), "s"(z4), "s"(z4) : "memory");
#endif
}

// pad config: row of 128 elems (256B) -> interval code 5; row of 32 elems (64B)
// -> code 3; pad_amount code 3 = 4 DWORDs = 16B = 8 elems  => strides 136 / 40.
#define TDM_W0_K ((1u << 16) | (1u << 20) | (5u << 22) | (3u << 25))
#define TDM_W0_V ((1u << 16) | (1u << 20) | (3u << 22) | (3u << 25))

// ---------------- prep: fp32 -> bf16 hi/lo ----------------
__global__ void k_split(const float* __restrict__ in, bf16_t* __restrict__ hi,
                        bf16_t* __restrict__ lo, int n) {
  int i = blockIdx.x * 256 + threadIdx.x;
  if (i < n) {
    float v = in[i];
    bf16_t h = (bf16_t)v;
    hi[i] = h;
    lo[i] = (bf16_t)(v - (float)h);
  }
}

// W is [K][N] row-major; write W^T hi/lo at out[(rowOff+n)*ldo + k].
__global__ void k_tsplit(const float* __restrict__ W, int K, int N,
                         bf16_t* __restrict__ oh, bf16_t* __restrict__ ol,
                         int ldo, int rowOff) {
  int idx = blockIdx.x * 256 + threadIdx.x;
  if (idx >= K * N) return;
  int k = idx / N, n = idx % N;
  float v = W[(size_t)k * N + n];
  size_t o = (size_t)(rowOff + n) * ldo + k;
  bf16_t h = (bf16_t)v;
  oh[o] = h;
  ol[o] = (bf16_t)(v - (float)h);
}

// ---------------- GEMM: C[M,N] = A[M,K] * B[K,N], B given transposed --------
// 256 threads = 8 waves (2x4 wave grid), block tile 128x128, BK=32.
// Register-prefetch pipeline: global fetch of slab t+1 overlaps WMMA on slab t.
template <bool SPLIT_OUT>
__global__ __launch_bounds__(256) void k_gemm(
    const bf16_t* __restrict__ Ah, const bf16_t* __restrict__ Al,
    const bf16_t* __restrict__ BTh, const bf16_t* __restrict__ BTl,
    float* __restrict__ C, bf16_t* __restrict__ Ch, bf16_t* __restrict__ Cl,
    int M, int N, int K) {
  alignas(16) __shared__ bf16_t sAh[128 * 40];
  alignas(16) __shared__ bf16_t sAl[128 * 40];
  alignas(16) __shared__ bf16_t sBh[128 * 40];
  alignas(16) __shared__ bf16_t sBl[128 * 40];

  const int tid = threadIdx.x;
  const int lane = tid & 31, wave = tid >> 5;
  const int wr = wave >> 2, wc = wave & 3;          // 2 x 4 waves
  const int l15 = lane & 15, kh8 = (lane >> 4) * 8; // fragment addressing
  const int hi8 = (lane >> 4) * 8;                  // C-frag row offset
  const int m0 = blockIdx.y * 128, n0 = blockIdx.x * 128;

  v8f acc[4][2];
#pragma unroll
  for (int i = 0; i < 4; ++i)
#pragma unroll
    for (int j = 0; j < 2; ++j)
#pragma unroll
      for (int v = 0; v < 8; ++v) acc[i][j][v] = 0.0f;

  const int r = tid >> 1;              // 128 rows, 2 threads per row
  const int ho = (tid & 1) * 16;       // 16 elems each
  const bf16_t* gAh = Ah + (size_t)(m0 + r) * K + ho;
  const bf16_t* gAl = Al + (size_t)(m0 + r) * K + ho;
  const bf16_t* gBh = BTh + (size_t)(n0 + r) * K + ho;
  const bf16_t* gBl = BTl + (size_t)(n0 + r) * K + ho;

  // prefetch slab 0 into registers
  v8bf ra0 = *(const v8bf*)(gAh),     ra1 = *(const v8bf*)(gAh + 8);
  v8bf rb0 = *(const v8bf*)(gAl),     rb1 = *(const v8bf*)(gAl + 8);
  v8bf rc0 = *(const v8bf*)(gBh),     rc1 = *(const v8bf*)(gBh + 8);
  v8bf rd0 = *(const v8bf*)(gBl),     rd1 = *(const v8bf*)(gBl + 8);

  for (int kk = 0; kk < K; kk += 32) {
    *(v8bf*)&sAh[r * 40 + ho]     = ra0;
    *(v8bf*)&sAh[r * 40 + ho + 8] = ra1;
    *(v8bf*)&sAl[r * 40 + ho]     = rb0;
    *(v8bf*)&sAl[r * 40 + ho + 8] = rb1;
    *(v8bf*)&sBh[r * 40 + ho]     = rc0;
    *(v8bf*)&sBh[r * 40 + ho + 8] = rc1;
    *(v8bf*)&sBl[r * 40 + ho]     = rd0;
    *(v8bf*)&sBl[r * 40 + ho + 8] = rd1;
    __syncthreads();

    if (kk + 32 < K) {  // overlap next-slab global fetch with WMMA below
      int k2 = kk + 32;
      ra0 = *(const v8bf*)(gAh + k2); ra1 = *(const v8bf*)(gAh + k2 + 8);
      rb0 = *(const v8bf*)(gAl + k2); rb1 = *(const v8bf*)(gAl + k2 + 8);
      rc0 = *(const v8bf*)(gBh + k2); rc1 = *(const v8bf*)(gBh + k2 + 8);
      rd0 = *(const v8bf*)(gBl + k2); rd1 = *(const v8bf*)(gBl + k2 + 8);
    }

    v16bf afh[4], afl[4], bfh[2], bfl[2];
#pragma unroll
    for (int i = 0; i < 4; ++i) {
      int ro = (wr * 64 + i * 16 + l15) * 40 + kh8;
      afh[i] = frag_ld(&sAh[ro]);
      afl[i] = frag_ld(&sAl[ro]);
    }
#pragma unroll
    for (int j = 0; j < 2; ++j) {
      int ro = (wc * 32 + j * 16 + l15) * 40 + kh8;
      bfh[j] = frag_ld(&sBh[ro]);
      bfl[j] = frag_ld(&sBl[ro]);
    }
#pragma unroll
    for (int i = 0; i < 4; ++i)
#pragma unroll
      for (int j = 0; j < 2; ++j) mma3(acc[i][j], afh[i], afl[i], bfh[j], bfl[j]);
    __syncthreads();
  }

#pragma unroll
  for (int i = 0; i < 4; ++i)
#pragma unroll
    for (int j = 0; j < 2; ++j)
#pragma unroll
      for (int v = 0; v < 8; ++v) {
        int row = m0 + wr * 64 + i * 16 + v + hi8;
        int col = n0 + wc * 32 + j * 16 + l15;
        float x = acc[i][j][v];
        size_t o = (size_t)row * N + col;
        if (SPLIT_OUT) {
          split_store(Ch, Cl, o, x);
        } else {
          C[o] = x;
        }
      }
}

// ---------------- RoPE + layout shuffle -------------------------------------
// qkv[ROWS][3072] (hi/lo) -> Q[B,NH,S,HD], K[B,NKV,S,HD] (rotated), Vt[B,NKV,HD,S]
__global__ void k_rope(const bf16_t* __restrict__ qkvh, const bf16_t* __restrict__ qkvl,
                       bf16_t* __restrict__ Qh, bf16_t* __restrict__ Ql,
                       bf16_t* __restrict__ Kh, bf16_t* __restrict__ Kl,
                       bf16_t* __restrict__ Vth, bf16_t* __restrict__ Vtl) {
  int idx = blockIdx.x * 256 + threadIdx.x;
  int row = idx / 1792;
  int u = idx % 1792;
  if (row >= ROWS) return;
  int b = row >> 11, s = row & 2047;
  const size_t rb = (size_t)row * NQKV;
  auto ld = [&](int col) -> float {
    return (float)qkvh[rb + col] + (float)qkvl[rb + col];
  };
  if (u < 1024) {                       // Q rotation pairs: 16 heads * 64 dims
    int hh = u >> 6, d = u & 63;
    float a = ld(hh * 128 + d), bb = ld(hh * 128 + d + 64);
    float ang = (float)s * __powf(10000.0f, -(float)d * (1.0f / 64.0f));
    float sn, cs;
    __sincosf(ang, &sn, &cs);
    size_t base = ((size_t)(b * NH + hh) * SEQ + s) * HD + d;
    split_store(Qh, Ql, base,      a * cs - bb * sn);
    split_store(Qh, Ql, base + 64, bb * cs + a * sn);
  } else if (u < 1280) {                // K rotation pairs: 4 kv heads * 64
    int u2 = u - 1024;
    int kvh = u2 >> 6, d = u2 & 63;
    float a = ld(2048 + kvh * 128 + d), bb = ld(2048 + kvh * 128 + d + 64);
    float ang = (float)s * __powf(10000.0f, -(float)d * (1.0f / 64.0f));
    float sn, cs;
    __sincosf(ang, &sn, &cs);
    size_t base = ((size_t)(b * NKV + kvh) * SEQ + s) * HD + d;
    split_store(Kh, Kl, base,      a * cs - bb * sn);
    split_store(Kh, Kl, base + 64, bb * cs + a * sn);
  } else {                              // V transpose copy: 512 elems
    int u2 = u - 1280;
    int kvh = u2 >> 7, d = u2 & 127;
    float vv = ld(2560 + u2);
    size_t o = ((size_t)(b * NKV + kvh) * HD + d) * SEQ + s;
    split_store(Vth, Vtl, o, vv);
  }
}

// ---------------- Flash attention (TDM-staged K/V, double-buffered) ---------
// Grid: (SEQ/64 q-tiles, BATCH*NH). 128 threads = 4 waves; wave w owns 16 rows.
__global__ __launch_bounds__(128) void k_flash(
    const bf16_t* __restrict__ Qh, const bf16_t* __restrict__ Ql,
    const bf16_t* __restrict__ Kh, const bf16_t* __restrict__ Kl,
    const bf16_t* __restrict__ Vth, const bf16_t* __restrict__ Vtl,
    bf16_t* __restrict__ Oh, bf16_t* __restrict__ Ol) {
  alignas(16) __shared__ bf16_t sKh[2][32 * 136];
  alignas(16) __shared__ bf16_t sKl[2][32 * 136];
  alignas(16) __shared__ bf16_t sVh[2][128 * 40];
  alignas(16) __shared__ bf16_t sVl[2][128 * 40];
  alignas(16) __shared__ bf16_t sPh[4 * 16 * 40];
  alignas(16) __shared__ bf16_t sPl[4 * 16 * 40];

  const int tid = threadIdx.x;
  const int lane = tid & 31, w = tid >> 5;
  const int l15 = lane & 15, kh8 = (lane >> 4) * 8, hi8 = (lane >> 4) * 8;
  const int q0 = blockIdx.x * 64;
  const int b = blockIdx.y >> 4, h = blockIdx.y & 15, kv = h >> 2;

  const bf16_t* Qbh = Qh + (size_t)(b * NH + h) * SEQ * HD;
  const bf16_t* Qbl = Ql + (size_t)(b * NH + h) * SEQ * HD;
  const bf16_t* Kbh = Kh + (size_t)(b * NKV + kv) * SEQ * HD;
  const bf16_t* Kbl = Kl + (size_t)(b * NKV + kv) * SEQ * HD;
  const bf16_t* Vbh = Vth + (size_t)(b * NKV + kv) * HD * SEQ;
  const bf16_t* Vbl = Vtl + (size_t)(b * NKV + kv) * HD * SEQ;

  // Resident Q fragments (16 rows x 128 dims, hi/lo)
  v16bf qfh[4], qfl[4];
  {
    int qr = q0 + w * 16 + l15;
    const bf16_t* ph = Qbh + (size_t)qr * HD + kh8;
    const bf16_t* pl = Qbl + (size_t)qr * HD + kh8;
#pragma unroll
    for (int ds = 0; ds < 4; ++ds) {
      qfh[ds] = frag_ld(ph + ds * 32);
      qfl[ds] = frag_ld(pl + ds * 32);
    }
  }

  v8f o[8];
#pragma unroll
  for (int j2 = 0; j2 < 8; ++j2)
#pragma unroll
    for (int v = 0; v < 8; ++v) o[j2][v] = 0.0f;
  float mrow[8], lrow[8];
#pragma unroll
  for (int v = 0; v < 8; ++v) { mrow[v] = -3.0e38f; lrow[v] = 0.0f; }

  const float scale = 0.08838834764831845f; // 1/sqrt(128)
  const int nk = q0 / 32 + 2;               // causal: k <= q0+63

  // TDM issue for k-tile t into LDS buffer t&1 (wave-uniform operands).
  auto issue_tile = [&](int t) {
    int nb = t & 1;
    unsigned long long koff = (unsigned long long)t * 32 * HD * 2;  // bytes
    unsigned long long voff = (unsigned long long)t * 32 * 2;       // bytes
    // K tile: 32 rows x 128 elems, row stride 128, pad 256B rows -> stride 136
    tdm_load2d(lds_off(&sKh[nb][0]), (unsigned long long)(size_t)Kbh + koff,
               TDM_W0_K, 128u, (unsigned)SEQ, 128u, 32u, 128u);
    tdm_load2d(lds_off(&sKl[nb][0]), (unsigned long long)(size_t)Kbl + koff,
               TDM_W0_K, 128u, (unsigned)SEQ, 128u, 32u, 128u);
    // V tile: 128 rows x 32 elems, row stride SEQ, pad 64B rows -> stride 40
    tdm_load2d(lds_off(&sVh[nb][0]), (unsigned long long)(size_t)Vbh + voff,
               TDM_W0_V, (unsigned)SEQ, 128u, 32u, 128u, (unsigned)SEQ);
    tdm_load2d(lds_off(&sVl[nb][0]), (unsigned long long)(size_t)Vbl + voff,
               TDM_W0_V, (unsigned)SEQ, 128u, 32u, 128u, (unsigned)SEQ);
  };

  if (tid < 32) issue_tile(0);  // one wave drives the DMA

  for (int t = 0; t < nk; ++t) {
    const int k0 = t * 32;
    const int buf = t & 1;
    if (tid < 32) __builtin_amdgcn_s_wait_tensorcnt(0);  // tile t landed in LDS
    __syncthreads();
    if (tid < 32 && t + 1 < nk) issue_tile(t + 1);       // DMA overlaps compute
    const bf16_t* bK_h = &sKh[buf][0];
    const bf16_t* bK_l = &sKl[buf][0];
    const bf16_t* bV_h = &sVh[buf][0];
    const bf16_t* bV_l = &sVl[buf][0];

    // ---- scores: S[16x32] = Q(16x128) * K^T ----
    v8f s0, s1;
#pragma unroll
    for (int v = 0; v < 8; ++v) { s0[v] = 0.0f; s1[v] = 0.0f; }
#pragma unroll
    for (int ds = 0; ds < 4; ++ds) {
      v16bf b0h = frag_ld(&bK_h[(l15) * 136 + ds * 32 + kh8]);
      v16bf b0l = frag_ld(&bK_l[(l15) * 136 + ds * 32 + kh8]);
      v16bf b1h = frag_ld(&bK_h[(16 + l15) * 136 + ds * 32 + kh8]);
      v16bf b1l = frag_ld(&bK_l[(16 + l15) * 136 + ds * 32 + kh8]);
      mma3(s0, qfh[ds], qfl[ds], b0h, b0l);
      mma3(s1, qfh[ds], qfl[ds], b1h, b1l);
    }

    // ---- scale + causal mask + online softmax (per-row stats in lane halves)
    float fac[8];
#pragma unroll
    for (int v = 0; v < 8; ++v) {
      int qr = q0 + w * 16 + v + hi8;
      float a0 = s0[v] * scale;
      if (k0 + l15 > qr) a0 = -3.0e38f;
      float a1 = s1[v] * scale;
      if (k0 + 16 + l15 > qr) a1 = -3.0e38f;
      float mt = fmaxf(a0, a1);
      mt = fmaxf(mt, __shfl_xor(mt, 8, 16));
      mt = fmaxf(mt, __shfl_xor(mt, 4, 16));
      mt = fmaxf(mt, __shfl_xor(mt, 2, 16));
      mt = fmaxf(mt, __shfl_xor(mt, 1, 16));
      float nm = fmaxf(mrow[v], mt);
      float f = __expf(mrow[v] - nm);
      mrow[v] = nm;
      a0 = __expf(a0 - nm);
      a1 = __expf(a1 - nm);
      float rs = a0 + a1;
      rs += __shfl_xor(rs, 8, 16);
      rs += __shfl_xor(rs, 4, 16);
      rs += __shfl_xor(rs, 2, 16);
      rs += __shfl_xor(rs, 1, 16);
      lrow[v] = lrow[v] * f + rs;
      fac[v] = f;
      s0[v] = a0;
      s1[v] = a1;
    }
#pragma unroll
    for (int j2 = 0; j2 < 8; ++j2)
#pragma unroll
      for (int v = 0; v < 8; ++v) o[j2][v] *= fac[v];

    // ---- route P (C-layout) through LDS to A-layout, hi/lo split ----
#pragma unroll
    for (int v = 0; v < 8; ++v) {
      int rw = v + hi8;
      int a = w * 640 + rw * 40;
      float p0 = s0[v];
      bf16_t h0 = (bf16_t)p0;
      sPh[a + l15] = h0;
      sPl[a + l15] = (bf16_t)(p0 - (float)h0);
      float p1 = s1[v];
      bf16_t h1 = (bf16_t)p1;
      sPh[a + 16 + l15] = h1;
      sPl[a + 16 + l15] = (bf16_t)(p1 - (float)h1);
    }
    asm volatile("s_wait_dscnt 0" ::: "memory"); // wave-local LDS RAW

    v16bf pfh = frag_ld(&sPh[w * 640 + l15 * 40 + kh8]);
    v16bf pfl = frag_ld(&sPl[w * 640 + l15 * 40 + kh8]);
#pragma unroll
    for (int j2 = 0; j2 < 8; ++j2) {
      v16bf vfh = frag_ld(&bV_h[(j2 * 16 + l15) * 40 + kh8]);
      v16bf vfl = frag_ld(&bV_l[(j2 * 16 + l15) * 40 + kh8]);
      mma3(o[j2], pfh, pfl, vfh, vfl);
    }
    __syncthreads();
  }

  // ---- epilogue: O /= l, write hi/lo bf16 directly for the out-proj GEMM ----
#pragma unroll
  for (int j2 = 0; j2 < 8; ++j2)
#pragma unroll
    for (int v = 0; v < 8; ++v) {
      int row = q0 + w * 16 + v + hi8;
      int col = h * HD + j2 * 16 + l15;
      size_t oi = (size_t)(b * SEQ + row) * (NH * HD) + col;
      split_store(Oh, Ol, oi, o[j2][v] / lrow[v]);
    }
}

// ---------------------------------------------------------------------------
extern "C" void kernel_launch(void* const* d_in, const int* in_sizes, int n_in,
                              void* d_out, int out_size, void* d_ws, size_t ws_size,
                              hipStream_t stream) {
  const float* x  = (const float*)d_in[0];
  const float* Wq = (const float*)d_in[1];
  const float* Wk = (const float*)d_in[2];
  const float* Wv = (const float*)d_in[3];
  const float* Wo = (const float*)d_in[4];

  const size_t NE_X    = (size_t)ROWS * HIDDEN;       // 8.39M
  const size_t NE_WQKV = (size_t)NQKV * HIDDEN;       // 6.29M
  const size_t NE_WO   = (size_t)HIDDEN * HIDDEN;     // 4.19M
  const size_t NE_QKV  = (size_t)ROWS * NQKV;         // 12.58M
  const size_t NE_Q    = (size_t)BATCH * NH * SEQ * HD;
  const size_t NE_KV   = (size_t)BATCH * NKV * SEQ * HD;

  char* w = (char*)d_ws;
  char* wend = w + ws_size;
  auto alloc = [&](size_t elems) -> bf16_t* {
    bf16_t* p = (bf16_t*)w;
    w += (elems * sizeof(bf16_t) + 255) & ~(size_t)255;
    return p;
  };
  bf16_t* xh     = alloc(NE_X);     bf16_t* xl     = alloc(NE_X);
  bf16_t* wqkvth = alloc(NE_WQKV);  bf16_t* wqkvtl = alloc(NE_WQKV);
  bf16_t* woth   = alloc(NE_WO);    bf16_t* wotl   = alloc(NE_WO);
  bf16_t* qkvh   = alloc(NE_QKV);   bf16_t* qkvl   = alloc(NE_QKV);
  bf16_t* qh     = alloc(NE_Q);     bf16_t* ql     = alloc(NE_Q);
  bf16_t* kh     = alloc(NE_KV);    bf16_t* kl     = alloc(NE_KV);
  bf16_t* vth    = alloc(NE_KV);    bf16_t* vtl    = alloc(NE_KV);
  if (w > wend) return; // ~160 MB required
  // attention output (bf16 hi/lo) reuses the x-split region (same size, x dead)
  bf16_t* ah = xh;
  bf16_t* al = xl;

  // 1) split x
  k_split<<<(int)(NE_X / 256), 256, 0, stream>>>(x, xh, xl, (int)NE_X);
  // 2) W^T splits: [Wq | Wk | Wv] rows 0..3071 of Wqkv^T; Wo^T
  k_tsplit<<<(2048 * 2048) / 256, 256, 0, stream>>>(Wq, HIDDEN, 2048, wqkvth, wqkvtl, HIDDEN, 0);
  k_tsplit<<<(2048 * 512) / 256, 256, 0, stream>>>(Wk, HIDDEN, 512, wqkvth, wqkvtl, HIDDEN, 2048);
  k_tsplit<<<(2048 * 512) / 256, 256, 0, stream>>>(Wv, HIDDEN, 512, wqkvth, wqkvtl, HIDDEN, 2560);
  k_tsplit<<<(2048 * 2048) / 256, 256, 0, stream>>>(Wo, 2048, HIDDEN, woth, wotl, 2048, 0);
  // 3) fused QKV projection (emits bf16 hi/lo qkv)
  k_gemm<true><<<dim3(NQKV / 128, ROWS / 128), 256, 0, stream>>>(
      xh, xl, wqkvth, wqkvtl, nullptr, qkvh, qkvl, ROWS, NQKV, HIDDEN);
  // 4) RoPE + layout (Q, K, V^T)
  k_rope<<<(ROWS * 1792) / 256, 256, 0, stream>>>(qkvh, qkvl, qh, ql, kh, kl, vth, vtl);
  // 5) causal flash attention (TDM-staged K/V)
  k_flash<<<dim3(SEQ / 64, BATCH * NH), 128, 0, stream>>>(qh, ql, kh, kl, vth, vtl, ah, al);
  // 6) output projection -> fp32 d_out
  k_gemm<false><<<dim3(HIDDEN / 128, ROWS / 128), 256, 0, stream>>>(
      ah, al, woth, wotl, (float*)d_out, nullptr, nullptr, ROWS, HIDDEN, 2048);
}
